// Word2VectorModelHierarchicalSoftmax_20023137534886
// MI455X (gfx1250) — matrix operand order repaired and checked by reading, not compiled, not run on hardware
//
#include <hip/hip_runtime.h>
#include <hip/hip_bf16.h>
#include <math.h>

typedef __attribute__((ext_vector_type(16))) __bf16 v16bf;
typedef __attribute__((ext_vector_type(4)))  __bf16 v4bf;
typedef __attribute__((ext_vector_type(8)))  float  v8f;

#define V_DIM 50000
#define D_DIM 128
#define B_DIM 2048
#define P_DIM 20
#define KSPLIT0 25024   // 782*32 ; second half = 24976 = 780*32 + 16 (tail)

// ---------------------------------------------------------------------------
// Kernel 0: one-shot convert W_emb [D,V] f32 -> bf16 (row-major preserved).
// 25.6MB read + 12.8MB write, amortized over the whole GEMM; removes 8
// v_cvt_pk_bf16_f32 + 2 global_load_b128 per k-step from the hot loop.
// ---------------------------------------------------------------------------
__global__ __launch_bounds__(256) void w2v_cvt_w_bf16(
    const float* __restrict__ W, __bf16* __restrict__ Wb)
{
  const size_t i = ((size_t)blockIdx.x * 256 + threadIdx.x) * 4;  // D*V % 4 == 0
  const float4 v = *(const float4*)(W + i);
  v4bf o;
  o[0] = (__bf16)v.x; o[1] = (__bf16)v.y; o[2] = (__bf16)v.z; o[3] = (__bf16)v.w;
  *(v4bf*)(Wb + i) = o;
}

// ---------------------------------------------------------------------------
// Kernel 1: X_part[kh] = A[2048,50000] x W_emb[128,50000]^T over K-half kh.
// BF16 WMMA 16x16x32, f32 accumulate. A converted fp32->bf16 on the fly
// (streamed once from HBM); B loaded directly as bf16 (L2-resident copy).
// 8 waves/block = 8 N-tiles sharing one 16-row M-tile; grid = (128, 2 k-halves)
// -> 2048 waves in flight. Unroll-4 keeps ~24MB of loads outstanding
// (~HBM bandwidth-delay product at 23.3 TB/s).
// ---------------------------------------------------------------------------
__global__ __launch_bounds__(256) void w2v_gemm_emb_bf16(
    const float* __restrict__ A, const __bf16* __restrict__ Wb,
    float* __restrict__ Xparts)
{
  const int lane    = threadIdx.x & 31;
  const int wvid    = threadIdx.x >> 5;      // 0..7 -> N tile
  const int mtile   = blockIdx.x;            // 0..127
  const int kh      = blockIdx.y;            // 0..1
  const int kstart  = kh ? KSPLIT0 : 0;
  const int kend    = kh ? V_DIM   : KSPLIT0;
  const int halfSel = lane >> 4;             // 0 for lanes 0-15, 1 for 16-31

  const int rowA = mtile * 16 + (lane & 15);     // A fragment: lane%16 = M
  const int rowW = wvid  * 16 + (lane & 15);     // B fragment: lane%16 = N
  const float*  __restrict__ aRow = A  + (size_t)rowA * V_DIM;
  const __bf16* __restrict__ wRow = Wb + (size_t)rowW * V_DIM;
  const int aOff = halfSel * 16;   // A: K group  {aOff..aOff+15}
  const int bOff = halfSel * 16;   // B: K group  {bOff..bOff+15}

  v8f acc = {};
  int k0 = kstart;
#pragma unroll 4
  for (; k0 + 32 <= kend; k0 += 32) {
    // prefetch the streamed A operand ~32 iterations (4KB) ahead into L2/L0
    __builtin_prefetch(aRow + k0 + aOff + 1024, 0, 0);

    const float4 a0 = *(const float4*)(aRow + k0 + aOff);
    const float4 a1 = *(const float4*)(aRow + k0 + aOff + 4);
    const float4 a2 = *(const float4*)(aRow + k0 + aOff + 8);
    const float4 a3 = *(const float4*)(aRow + k0 + aOff + 12);
    const v16bf bfr = *(const v16bf*)(wRow + k0 + bOff);   // 32B aligned

    v16bf af;
    af[0]=(__bf16)a0.x;  af[1]=(__bf16)a0.y;  af[2]=(__bf16)a0.z;  af[3]=(__bf16)a0.w;
    af[4]=(__bf16)a1.x;  af[5]=(__bf16)a1.y;  af[6]=(__bf16)a1.z;  af[7]=(__bf16)a1.w;
    af[8]=(__bf16)a2.x;  af[9]=(__bf16)a2.y;  af[10]=(__bf16)a2.z; af[11]=(__bf16)a2.w;
    af[12]=(__bf16)a3.x; af[13]=(__bf16)a3.y; af[14]=(__bf16)a3.z; af[15]=(__bf16)a3.w;

    acc = __builtin_amdgcn_wmma_f32_16x16x32_bf16(
        false, af, false, bfr, (short)0, acc, false, false);
  }

  // K tail: exactly 16 remaining (only in the second K-half). Zero-pad K>=16.
  if (k0 < kend) {
    v16bf af = {}, bfr = {};
    if (lane < 16) {
    const float4 a0 = *(const float4*)(aRow + k0 + aOff);
    const float4 a1 = *(const float4*)(aRow + k0 + aOff + 4);
    const float4 a2 = *(const float4*)(aRow + k0 + aOff + 8);
    const float4 a3 = *(const float4*)(aRow + k0 + aOff + 12);
    af[0]=(__bf16)a0.x; af[1]=(__bf16)a0.y; af[2]=(__bf16)a0.z; af[3]=(__bf16)a0.w;
    af[4]=(__bf16)a1.x; af[5]=(__bf16)a1.y; af[6]=(__bf16)a1.z; af[7]=(__bf16)a1.w;
    af[8]=(__bf16)a2.x; af[9]=(__bf16)a2.y; af[10]=(__bf16)a2.z; af[11]=(__bf16)a2.w;
    af[12]=(__bf16)a3.x; af[13]=(__bf16)a3.y; af[14]=(__bf16)a3.z; af[15]=(__bf16)a3.w;
    }
    if (lane < 16) {   // lanes 16-31 cover K=16..31 -> all zero
      bfr = *(const v16bf*)(wRow + k0);   // ends exactly at row end
    }
    acc = __builtin_amdgcn_wmma_f32_16x16x32_bf16(
        false, af, false, bfr, (short)0, acc, false, false);
  }

  // C/D layout: VGPR r -> M = r + 8*halfSel, N = lane%16
  float* __restrict__ out = Xparts + (size_t)kh * (B_DIM * D_DIM);
  const int mBase = mtile * 16 + halfSel * 8;
  const int nOut  = wvid * 16 + (lane & 15);
#pragma unroll
  for (int r = 0; r < 8; ++r)
    out[(size_t)(mBase + r) * D_DIM + nOut] = acc[r];
}

// ---------------------------------------------------------------------------
// Kernel 2: per-sample hierarchical-softmax BCE. Block = sample, 4 waves.
// Sums the two K-half GEMM partials while loading x[b,:].
// ---------------------------------------------------------------------------
__global__ __launch_bounds__(128) void w2v_hs_loss(
    const float* __restrict__ Xparts, const float* __restrict__ Wc,
    const int* __restrict__ nodes, const float* __restrict__ codes,
    const float* __restrict__ mask, float* __restrict__ perSample)
{
  const int b    = blockIdx.x;
  const int lane = threadIdx.x & 31;
  const int wvid = threadIdx.x >> 5;     // 0..3
  __shared__ float sLoss[4], sMask[4];

  const float4 x0 = *(const float4*)(Xparts + (size_t)b * D_DIM + lane * 4);
  const float4 x1 = *(const float4*)(Xparts + (size_t)(B_DIM + b) * D_DIM + lane * 4);
  const float xv0 = x0.x + x1.x, xv1 = x0.y + x1.y;
  const float xv2 = x0.z + x1.z, xv3 = x0.w + x1.w;

  float lossAcc = 0.0f, maskAcc = 0.0f;
  for (int p = wvid; p < P_DIM; p += 4) {
    const int node = nodes[b * P_DIM + p];
    const float4 wv = *(const float4*)(Wc + (size_t)node * D_DIM + lane * 4);
    float d = xv0 * wv.x + xv1 * wv.y + xv2 * wv.z + xv3 * wv.w;
#pragma unroll
    for (int off = 16; off > 0; off >>= 1) d += __shfl_xor(d, off, 32);
    const float t = codes[b * P_DIM + p];
    const float m = mask[b * P_DIM + p];
    const float bce = fmaxf(d, 0.0f) - d * t + log1pf(expf(-fabsf(d)));
    lossAcc += bce * m;
    maskAcc += m;
  }
  if (lane == 0) { sLoss[wvid] = lossAcc; sMask[wvid] = maskAcc; }
  __syncthreads();
  if (threadIdx.x == 0) {
    const float L = sLoss[0] + sLoss[1] + sLoss[2] + sLoss[3];
    const float M = sMask[0] + sMask[1] + sMask[2] + sMask[3];
    perSample[b] = L / M;
  }
}

// ---------------------------------------------------------------------------
// Kernel 3: mean over the 2048 per-sample losses -> d_out[0].
// ---------------------------------------------------------------------------
__global__ __launch_bounds__(256) void w2v_reduce_mean(
    const float* __restrict__ perSample, float* __restrict__ out)
{
  __shared__ float s[256];
  float acc = 0.0f;
  for (int i = threadIdx.x; i < B_DIM; i += 256) acc += perSample[i];
  s[threadIdx.x] = acc;
  __syncthreads();
  for (int stride = 128; stride > 0; stride >>= 1) {
    if ((int)threadIdx.x < stride) s[threadIdx.x] += s[threadIdx.x + stride];
    __syncthreads();
  }
  if (threadIdx.x == 0) out[0] = s[0] * (1.0f / (float)B_DIM);
}

// ---------------------------------------------------------------------------
extern "C" void kernel_launch(void* const* d_in, const int* in_sizes, int n_in,
                              void* d_out, int out_size, void* d_ws, size_t ws_size,
                              hipStream_t stream) {
  (void)in_sizes; (void)n_in; (void)out_size; (void)ws_size;
  const float* A     = (const float*)d_in[0];   // inputs_vector [B,V]
  const float* Wemb  = (const float*)d_in[1];   // W_emb [D,V]
  const float* Wcls  = (const float*)d_in[2];   // W_cls [V,D]
  const int*   nodes = (const int*)  d_in[3];   // path_nodes_indices [B,P]
  const float* codes = (const float*)d_in[4];   // huffman_codes [B,P]
  const float* mask  = (const float*)d_in[5];   // mask [B,P]

  // scratch layout
  const size_t WB_BYTES = (size_t)D_DIM * V_DIM * sizeof(__bf16);    // 12.8 MB
  const size_t XP_BYTES = (size_t)2 * B_DIM * D_DIM * sizeof(float); // 2 MB
  __bf16* Wb        = (__bf16*)d_ws;
  float*  Xparts    = (float*)((char*)d_ws + WB_BYTES);
  float*  perSample = (float*)((char*)d_ws + WB_BYTES + XP_BYTES);

  const int cvtBlocks = (D_DIM * V_DIM) / (256 * 4);  // 6.4M elems, 4/thread
  w2v_cvt_w_bf16<<<cvtBlocks, 256, 0, stream>>>(Wemb, Wb);
  w2v_gemm_emb_bf16<<<dim3(128, 2), 256, 0, stream>>>(A, Wb, Xparts);
  w2v_hs_loss<<<B_DIM, 128, 0, stream>>>(Xparts, Wcls, nodes, codes, mask, perSample);
  w2v_reduce_mean<<<1, 256, 0, stream>>>(perSample, (float*)d_out);
}
